// PerlinSelfAttention_45337674777151
// MI455X (gfx1250) — matrix operand order repaired and checked by reading, not compiled, not run on hardware
//
#include <hip/hip_runtime.h>

// ---------------- problem constants ----------------
#define Bq_ 2
#define Sq_ 1024
#define Hq_ 1024
#define NHq_ 16
#define HDq_ 64
#define Rq_ 32
#define SCALEq_ 0.125f   // 1/sqrt(64)

typedef __attribute__((ext_vector_type(16))) __bf16    v16bf;
typedef __attribute__((ext_vector_type(8)))  float     v8f;
typedef __attribute__((ext_vector_type(8)))  unsigned  v8u;
typedef __attribute__((ext_vector_type(4)))  int       v4i;
typedef __attribute__((ext_vector_type(2)))  int       v2i;

static constexpr size_t PROJN = (size_t)Bq_ * NHq_ * Sq_ * HDq_;

// workspace byte offsets
static constexpr size_t OFF_XBF  = 0;
static constexpr size_t SZ_XBF   = (size_t)Bq_ * Sq_ * Hq_ * 2;
static constexpr size_t OFF_WEFF = OFF_XBF + SZ_XBF;
static constexpr size_t SZ_WEFF  = (size_t)8 * Hq_ * Hq_ * 2;
static constexpr size_t OFF_PROJ = OFF_WEFF + SZ_WEFF;
static constexpr size_t SZ_PROJ  = (size_t)8 * PROJN * 2;
static constexpr size_t OFF_KTH  = OFF_PROJ + SZ_PROJ;

// ---------------- optional gfx1250 async global->LDS path ----------------
#if defined(__has_builtin)
#if __has_builtin(__builtin_amdgcn_global_load_async_to_lds_b128) && \
    __has_builtin(__builtin_amdgcn_global_load_async_to_lds_b64) && \
    __has_builtin(__builtin_amdgcn_s_wait_asynccnt)
#define HAVE_ASYNC 1
#endif
#endif
#ifndef HAVE_ASYNC
#define HAVE_ASYNC 0
#endif

#if HAVE_ASYNC
__device__ inline void async_cp128(void* lds, const void* g) {
  __builtin_amdgcn_global_load_async_to_lds_b128(
      (__attribute__((address_space(1))) v4i*)g,
      (__attribute__((address_space(3))) v4i*)lds, 0, 0);
}
__device__ inline void async_cp64(void* lds, const void* g) {
  __builtin_amdgcn_global_load_async_to_lds_b64(
      (__attribute__((address_space(1))) v2i*)g,
      (__attribute__((address_space(3))) v2i*)lds, 0, 0);
}
__device__ inline void async_join() { __builtin_amdgcn_s_wait_asynccnt(0); }
#else
__device__ inline void async_cp128(void* lds, const void* g) {
  *(uint4*)lds = *(const uint4*)g;
}
__device__ inline void async_cp64(void* lds, const void* g) {
  *(uint2*)lds = *(const uint2*)g;
}
__device__ inline void async_join() {}
#endif

// ---------------- WMMA helpers ----------------
__device__ inline v8f wmma_bf(v16bf a, v16bf b, v8f c) {
  return __builtin_amdgcn_wmma_f32_16x16x32_bf16(false, a, false, b, (short)0, c, false, false);
}

// A fragment: 16x32 bf16 row-major tile; 2x contiguous 16B chunks per lane
__device__ inline v16bf load_frag_a_u(const __bf16* lds, int row0, int k0, int ld) {
  int lane = threadIdx.x & 31;
  int hf = lane >> 4;
  int m = lane & 15;
  const unsigned* p0 = (const unsigned*)__builtin_assume_aligned(
      (const void*)(lds + (size_t)(row0 + m) * ld + k0 + 16 * hf), 16);
  const unsigned* p1 = (const unsigned*)__builtin_assume_aligned(
      (const void*)(lds + (size_t)(row0 + m) * ld + k0 + 16 * hf + 8), 16);
  v8u u;
#pragma unroll
  for (int i = 0; i < 4; ++i) u[i] = p0[i];
#pragma unroll
  for (int i = 0; i < 4; ++i) u[4 + i] = p1[i];
  return __builtin_bit_cast(v16bf, u);
}

// B fragment from fragment-major tile: tileF[n*36 + kp], per-lane 8 contiguous uints
__device__ inline v16bf load_frag_bf(const unsigned* tileF, int n0, int kp0) {
  int lane = threadIdx.x & 31;
  int hf = lane >> 4;
  int n = lane & 15;
  const unsigned* p = (const unsigned*)__builtin_assume_aligned(
      (const void*)(tileF + (size_t)(n0 + n) * 36 + kp0 + 8 * hf), 16);
  v8u u;
#pragma unroll
  for (int i = 0; i < 8; ++i) u[i] = p[i];
  return __builtin_bit_cast(v16bf, u);
}

__device__ inline unsigned pack_lo(unsigned a, unsigned b) { return (a & 0xFFFFu) | (b << 16); }
__device__ inline unsigned pack_hi(unsigned a, unsigned b) { return (a >> 16) | (b & 0xFFFF0000u); }

__device__ inline unsigned pack2bf(float lo, float hi) {
  unsigned short l = __builtin_bit_cast(unsigned short, (__bf16)lo);
  unsigned short h = __builtin_bit_cast(unsigned short, (__bf16)hi);
  return (unsigned)l | ((unsigned)h << 16);
}

__device__ inline unsigned f2ord(float f) {
  unsigned u = __float_as_uint(f);
  return (u & 0x80000000u) ? ~u : (u | 0x80000000u);
}
__device__ inline float ord2f(unsigned u) {
  unsigned b = (u & 0x80000000u) ? (u & 0x7FFFFFFFu) : ~u;
  return __uint_as_float(b);
}

// ---------------- kernel 1: f32 -> bf16 convert of x ----------------
__global__ void convert_x_kernel(const float* __restrict__ x, unsigned* __restrict__ xbf_u, int n2) {
  for (int i = blockIdx.x * blockDim.x + threadIdx.x; i < n2; i += gridDim.x * blockDim.x)
    xbf_u[i] = pack2bf(x[2 * i], x[2 * i + 1]);
}

// ---------------- kernel 2: W_eff[n] = W_base[n] + A_n @ B_n (rank-32) ----------------
struct WeffArgs {
  const float* W[3];
  const float* A[8];
  const float* Bm[8];
};

__global__ void build_weff_kernel(WeffArgs args, unsigned* __restrict__ weff_u) {
  constexpr int baseIdx[8] = {0, 1, 2, 0, 1, 0, 1, 2};
  size_t idx = (size_t)blockIdx.x * blockDim.x + threadIdx.x;
  int n = (int)(idx >> 19);
  int rem = (int)(idx & 0x7FFFF);
  int i = rem >> 9;
  int j = (rem & 511) * 2;
  const float* Wb = args.W[baseIdx[n]];
  float acc0 = Wb[(size_t)i * Hq_ + j];
  float acc1 = Wb[(size_t)i * Hq_ + j + 1];
  const float* A = args.A[n];
  const float* Bm = args.Bm[n];
#pragma unroll 8
  for (int r = 0; r < Rq_; ++r) {
    float av = A[(size_t)i * Rq_ + r];
    acc0 += av * Bm[(size_t)r * Hq_ + j];
    acc1 += av * Bm[(size_t)r * Hq_ + j + 1];
  }
  weff_u[idx] = pack2bf(acc0, acc1);
}

// ---------------- kernel 3: proj[n] = x @ W_eff[n] + bias ----------------
__global__ __launch_bounds__(128) void proj_gemm_kernel(const __bf16* __restrict__ xbf,
                                                        const __bf16* __restrict__ weff,
                                                        const float* __restrict__ bq,
                                                        const float* __restrict__ bk,
                                                        const float* __restrict__ bvv,
                                                        __bf16* __restrict__ proj) {
  __shared__ __bf16   XT[64 * 40];    // 64 rows x 32 K (pitch 80B)
  __shared__ unsigned WTp[64 * 36];   // fragment-major: [n][kp], kp<16 used
  constexpr int baseIdx[8] = {0, 1, 2, 0, 1, 0, 1, 2};
  const int n = blockIdx.z;
  const int row0 = blockIdx.y * 64;
  const int col0 = blockIdx.x * 64;
  const __bf16* W = weff + (size_t)n * Hq_ * Hq_;
  const float* bias = (baseIdx[n] == 0) ? bq : ((baseIdx[n] == 1) ? bk : bvv);
  const int wave = threadIdx.x >> 5;
  const int lane = threadIdx.x & 31;
  const int hf = lane >> 4;
  const int nn = lane & 15;

  v8f acc[4] = {};
  for (int kc = 0; kc < Hq_; kc += 32) {
    __syncthreads();
    // x tile: async 128-bit copies (2 x 128 lanes)
#pragma unroll
    for (int it = 0; it < 2; ++it) {
      int idx = threadIdx.x + it * 128;
      int r = idx >> 2, c4 = idx & 3;
      async_cp128(XT + r * 40 + 8 * c4, xbf + (size_t)(row0 + r) * Hq_ + kc + 8 * c4);
    }
    // W tile: K-pair pack into fragment-major layout
#pragma unroll
    for (int it = 0; it < 4; ++it) {
      int idx = threadIdx.x + it * 128;
      int r = idx >> 5, c2 = idx & 31;   // r = kp, n = 2*c2(+1)
      unsigned a = *(const unsigned*)(W + (size_t)(kc + 2 * r) * Hq_ + col0 + 2 * c2);
      unsigned b = *(const unsigned*)(W + (size_t)(kc + 2 * r + 1) * Hq_ + col0 + 2 * c2);
      WTp[(2 * c2) * 36 + r]     = pack_lo(a, b);
      WTp[(2 * c2 + 1) * 36 + r] = pack_hi(a, b);
    }
    async_join();
    __syncthreads();
    v16bf a = load_frag_a_u(XT, wave * 16, 0, 40);
#pragma unroll
    for (int t = 0; t < 4; ++t)
      acc[t] = wmma_bf(a, load_frag_bf(WTp, t * 16, 0), acc[t]);
  }
#pragma unroll
  for (int t = 0; t < 4; ++t) {
    int col = col0 + t * 16 + nn;
    float bvad = bias[col];
    int head = col >> 6, d = col & 63;
#pragma unroll
    for (int e = 0; e < 8; ++e) {
      int row = row0 + wave * 16 + e + 8 * hf;
      int b = row >> 10, s = row & 1023;
      float val = acc[t][e] + bvad;
      proj[(size_t)n * PROJN + ((size_t)(b * NHq_ + head) * Sq_ + s) * HDq_ + d] = (__bf16)val;
    }
  }
}

// ---------------- kernel 4: per-row top-k threshold of sc_est ----------------
__global__ __launch_bounds__(128) void topk_kernel(const __bf16* __restrict__ proj,
                                                   const float* __restrict__ mask,
                                                   const int* __restrict__ topk_p,
                                                   float* __restrict__ kth) {
  extern __shared__ char smem[];
  float*    SC  = (float*)smem;                      // 32 x 1024 f32   (131072 B)
  __bf16*   QS  = (__bf16*)(smem + 131072);          // 32 x 72 bf16    (4608 B)
  unsigned* KSp = (unsigned*)(smem + 135680);        // frag-major [128 keys][36] (18432 B)
  const int bh = blockIdx.x;
  const int qt = blockIdx.y;
  const int b = bh >> 4;
  const __bf16* qs  = proj + 3 * PROJN + (size_t)bh * Sq_ * HDq_;
  const __bf16* ks_ = proj + 4 * PROJN + (size_t)bh * Sq_ * HDq_;
  const int wave = threadIdx.x >> 5;
  const int lane = threadIdx.x & 31;
  const int hf = lane >> 4;
  const int nn = lane & 15;

#pragma unroll
  for (int it = 0; it < 2; ++it) {
    int idx = threadIdx.x + it * 128;
    int r = idx >> 3, c4 = idx & 7;
    async_cp128(QS + r * 72 + 8 * c4, qs + (size_t)(qt * 32 + r) * HDq_ + 8 * c4);
  }
  for (int kb = 0; kb < 8; ++kb) {
    __syncthreads();
    // ks block [128 keys][64 d] -> frag-major [key][dpair]: straight contiguous copy
#pragma unroll
    for (int it = 0; it < 16; ++it) {
      int idx = threadIdx.x + it * 128;
      int key = idx >> 4, c2 = idx & 15;
      async_cp64(KSp + key * 36 + 2 * c2,
                 ks_ + (size_t)(kb * 128 + key) * HDq_ + 4 * c2);
    }
    async_join();
    __syncthreads();
#pragma unroll
    for (int it = 0; it < 4; ++it) {
      int tt = wave * 4 + it;
      int rb = tt >> 3, kt = tt & 7;
      v16bf a0 = load_frag_a_u(QS, rb * 16, 0, 72);
      v16bf a1 = load_frag_a_u(QS, rb * 16, 32, 72);
      v8f acc = {};
      acc = wmma_bf(a0, load_frag_bf(KSp, kt * 16, 0), acc);
      acc = wmma_bf(a1, load_frag_bf(KSp, kt * 16, 16), acc);
      int col = kb * 128 + kt * 16 + nn;
      float mv = mask[(size_t)b * Sq_ + col];
#pragma unroll
      for (int e = 0; e < 8; ++e) {
        int r = rb * 16 + e + 8 * hf;
        SC[r * 1024 + col] = acc[e] * SCALEq_ + mv;
      }
    }
  }
  __syncthreads();
  const int K = *topk_p;
  for (int r = wave * 8; r < wave * 8 + 8; ++r) {
    unsigned T = 0;
    for (int bit = 31; bit >= 0; --bit) {
      unsigned cand = T | (1u << bit);
      int cnt = 0;
      for (int j = lane; j < 1024; j += 32)
        cnt += (f2ord(SC[r * 1024 + j]) >= cand) ? 1 : 0;
      for (int m = 16; m >= 1; m >>= 1) cnt += __shfl_xor(cnt, m, 32);
      if (cnt >= K) T = cand;
    }
    if (lane == 0) kth[(size_t)bh * Sq_ + qt * 32 + r] = ord2f(T);
  }
}

// ---------------- online softmax (C-layout, 16-lane-half row reductions) ----------------
__device__ inline void online_update(v8f (&s)[4], float (&m)[8], float (&l)[8],
                                     v8f (&acc)[4], __bf16* P, int hf, int nn) {
#pragma unroll
  for (int e = 0; e < 8; ++e) {
    float mx = fmaxf(fmaxf(s[0][e], s[1][e]), fmaxf(s[2][e], s[3][e]));
    mx = fmaxf(mx, __shfl_xor(mx, 1, 32));
    mx = fmaxf(mx, __shfl_xor(mx, 2, 32));
    mx = fmaxf(mx, __shfl_xor(mx, 4, 32));
    mx = fmaxf(mx, __shfl_xor(mx, 8, 32));
    float mn = fmaxf(m[e], mx);
    float alpha = __expf(m[e] - mn);
    m[e] = mn;
    float rs = 0.f;
    int r = e + 8 * hf;
#pragma unroll
    for (int t = 0; t < 4; ++t) {
      float p = __expf(s[t][e] - mn);
      rs += p;
      acc[t][e] *= alpha;
      P[r * 72 + t * 16 + nn] = (__bf16)p;
    }
    rs += __shfl_xor(rs, 1, 32);
    rs += __shfl_xor(rs, 2, 32);
    rs += __shfl_xor(rs, 4, 32);
    rs += __shfl_xor(rs, 8, 32);
    l[e] = l[e] * alpha + rs;
  }
}

// ---------------- kernel 5: fused dual-branch sparse attention ----------------
__global__ __launch_bounds__(128) void attn_kernel(const __bf16* __restrict__ proj,
                                                   const float* __restrict__ mask,
                                                   const float* __restrict__ kth,
                                                   float* __restrict__ out) {
  extern __shared__ char smem[];
  __bf16*   QQ  = (__bf16*)smem;                 // 64x72 bf16 (9216 B each)
  __bf16*   QS2 = (__bf16*)(smem + 9216);
  __bf16*   QA  = (__bf16*)(smem + 18432);
  unsigned* KAp = (unsigned*)(smem + 27648);     // frag-major [64 keys][36] (9216 B each)
  unsigned* KSp = (unsigned*)(smem + 36864);
  unsigned* KKp = (unsigned*)(smem + 46080);
  unsigned* VVp = (unsigned*)(smem + 55296);     // frag-major [64 d][36] keypair-packed
  unsigned* VAp = (unsigned*)(smem + 64512);
  __bf16*   PB  = (__bf16*)(smem + 73728);       // 4 waves x 2 x 16x72 (18432 B)

  const int bh = blockIdx.x;
  const int qt = blockIdx.y;
  const int b = bh >> 4, head = bh & 15;
  const size_t hoff = (size_t)bh * Sq_ * HDq_;
  const __bf16* pq  = proj + 0 * PROJN + hoff;
  const __bf16* pk  = proj + 1 * PROJN + hoff;
  const __bf16* pv  = proj + 2 * PROJN + hoff;
  const __bf16* pqs = proj + 3 * PROJN + hoff;
  const __bf16* pks = proj + 4 * PROJN + hoff;
  const __bf16* pqa = proj + 5 * PROJN + hoff;
  const __bf16* pka = proj + 6 * PROJN + hoff;
  const __bf16* pva = proj + 7 * PROJN + hoff;

  const int wave = threadIdx.x >> 5;
  const int lane = threadIdx.x & 31;
  const int hf = lane >> 4;
  const int nn = lane & 15;
  const int row0 = wave * 16;

  // persistent q-side tiles: async 128-bit copies
#pragma unroll
  for (int it = 0; it < 4; ++it) {
    int idx = threadIdx.x + it * 128;
    int r = idx >> 3, c4 = idx & 7;
    size_t src = (size_t)(qt * 64 + r) * HDq_ + 8 * c4;
    async_cp128(QQ  + r * 72 + 8 * c4, pq  + src);
    async_cp128(QS2 + r * 72 + 8 * c4, pqs + src);
    async_cp128(QA  + r * 72 + 8 * c4, pqa + src);
  }
  async_join();
  __syncthreads();

  v16bf aQ0 = load_frag_a_u(QQ,  row0, 0, 72), aQ1 = load_frag_a_u(QQ,  row0, 32, 72);
  v16bf aS0 = load_frag_a_u(QS2, row0, 0, 72), aS1 = load_frag_a_u(QS2, row0, 32, 72);
  v16bf aA0 = load_frag_a_u(QA,  row0, 0, 72), aA1 = load_frag_a_u(QA,  row0, 32, 72);

  float kthv[8];
#pragma unroll
  for (int e = 0; e < 8; ++e) {
    int s_abs = qt * 64 + row0 + e + 8 * hf;
    kthv[e] = kth[(size_t)bh * Sq_ + s_abs];
  }

  v8f accA[4] = {}, accB[4] = {};
  float mA[8], lA[8], mB[8], lB[8];
#pragma unroll
  for (int e = 0; e < 8; ++e) { mA[e] = mB[e] = -3.0e38f; lA[e] = lB[e] = 0.f; }

  __bf16* PE  = PB + (size_t)wave * (2 * 16 * 72);
  __bf16* PQv = PE + 16 * 72;

  for (int kb = 0; kb < 16; ++kb) {
    __syncthreads();
    // k-side tiles: frag-major == row-major [key][dpair] -> contiguous async b64 copies
#pragma unroll
    for (int it = 0; it < 8; ++it) {
      int idx = threadIdx.x + it * 128;
      int key = idx >> 4, c2 = idx & 15;
      size_t src = (size_t)(kb * 64 + key) * HDq_ + 4 * c2;
      int o = key * 36 + 2 * c2;
      async_cp64(KKp + o, pk  + src);
      async_cp64(KSp + o, pks + src);
      async_cp64(KAp + o, pka + src);
    }
    // v-side tiles: keypair pack into frag-major [d][kp]
#pragma unroll
    for (int it = 0; it < 8; ++it) {
      int idx = threadIdx.x + it * 128;
      int kp = idx >> 5, c = idx & 31;
      size_t s0 = (size_t)(kb * 64 + 2 * kp) * HDq_ + 2 * c;
      size_t s1 = s0 + HDq_;
      unsigned a, bb;
      a = *(const unsigned*)(pv + s0); bb = *(const unsigned*)(pv + s1);
      VVp[(2 * c) * 36 + kp]     = pack_lo(a, bb);
      VVp[(2 * c + 1) * 36 + kp] = pack_hi(a, bb);
      a = *(const unsigned*)(pva + s0); bb = *(const unsigned*)(pva + s1);
      VAp[(2 * c) * 36 + kp]     = pack_lo(a, bb);
      VAp[(2 * c + 1) * 36 + kp] = pack_hi(a, bb);
    }
    async_join();
    __syncthreads();

    // prefetch next key block into cache while computing (global_prefetch_b8)
    if (kb + 1 < 16 && threadIdx.x < 64) {
      size_t nsrc = (size_t)(kb * 64 + 64 + threadIdx.x) * HDq_;
      __builtin_prefetch(pk + nsrc, 0, 0);
      __builtin_prefetch(pv + nsrc, 0, 0);
      __builtin_prefetch(pka + nsrc, 0, 0);
    }

    v8f sE[4], sQ[4];
#pragma unroll
    for (int t = 0; t < 4; ++t) {
      v8f e_ = {}; v8f c_ = {}; v8f q_ = {};
      e_ = wmma_bf(aA0, load_frag_bf(KAp, t * 16, 0),  e_);
      e_ = wmma_bf(aA1, load_frag_bf(KAp, t * 16, 16), e_);
      c_ = wmma_bf(aS0, load_frag_bf(KSp, t * 16, 0),  c_);
      c_ = wmma_bf(aS1, load_frag_bf(KSp, t * 16, 16), c_);
      q_ = wmma_bf(aQ0, load_frag_bf(KKp, t * 16, 0),  q_);
      q_ = wmma_bf(aQ1, load_frag_bf(KKp, t * 16, 16), q_);
      int col = kb * 64 + t * 16 + nn;
      float mv = mask[(size_t)b * Sq_ + col];
#pragma unroll
      for (int e = 0; e < 8; ++e) {
        float se = e_[e] * SCALEq_ + mv;
        float sc = c_[e] * SCALEq_ + mv;
        float sq = (sc >= kthv[e]) ? (q_[e] * SCALEq_ + mv) : -1.0e9f;
        sE[t][e] = se;
        sQ[t][e] = sq;
      }
    }

    online_update(sE, mA, lA, accA, PE,  hf, nn);
    online_update(sQ, mB, lB, accB, PQv, hf, nn);
    __syncthreads();

    v16bf pA0 = load_frag_a_u(PE,  0, 0, 72), pA1 = load_frag_a_u(PE,  0, 32, 72);
    v16bf pB0 = load_frag_a_u(PQv, 0, 0, 72), pB1 = load_frag_a_u(PQv, 0, 32, 72);
#pragma unroll
    for (int t = 0; t < 4; ++t) {
      accA[t] = wmma_bf(pA0, load_frag_bf(VAp, t * 16, 0),  accA[t]);
      accA[t] = wmma_bf(pA1, load_frag_bf(VAp, t * 16, 16), accA[t]);
      accB[t] = wmma_bf(pB0, load_frag_bf(VVp, t * 16, 0),  accB[t]);
      accB[t] = wmma_bf(pB1, load_frag_bf(VVp, t * 16, 16), accB[t]);
    }
  }

#pragma unroll
  for (int t = 0; t < 4; ++t) {
#pragma unroll
    for (int e = 0; e < 8; ++e) {
      int s_abs = qt * 64 + row0 + e + 8 * hf;
      int d = t * 16 + nn;
      float val = 0.5f * (accA[t][e] / lA[e] + accB[t][e] / lB[e]);
      out[((size_t)(b * Sq_ + s_abs)) * Hq_ + head * 64 + d] = val;
    }
  }
}

// ---------------- host launcher ----------------
extern "C" void kernel_launch(void* const* d_in, const int* in_sizes, int n_in,
                              void* d_out, int out_size, void* d_ws, size_t ws_size,
                              hipStream_t stream) {
  (void)in_sizes; (void)n_in; (void)out_size; (void)ws_size;
  const float* x    = (const float*)d_in[0];
  const float* mask = (const float*)d_in[1];
  const float* bq   = (const float*)d_in[3];
  const float* bk   = (const float*)d_in[5];
  const float* bv   = (const float*)d_in[7];
  const int* topk_p = (const int*)d_in[24];

  char* ws = (char*)d_ws;
  __bf16* xbf  = (__bf16*)(ws + OFF_XBF);
  __bf16* weff = (__bf16*)(ws + OFF_WEFF);
  __bf16* proj = (__bf16*)(ws + OFF_PROJ);
  float*  kth  = (float*)(ws + OFF_KTH);

  convert_x_kernel<<<1024, 256, 0, stream>>>(x, (unsigned*)xbf, (Bq_ * Sq_ * Hq_) / 2);

  WeffArgs wa;
  wa.W[0] = (const float*)d_in[2];
  wa.W[1] = (const float*)d_in[4];
  wa.W[2] = (const float*)d_in[6];
  for (int j = 0; j < 8; ++j) {
    wa.A[j]  = (const float*)d_in[8 + 2 * j];
    wa.Bm[j] = (const float*)d_in[9 + 2 * j];
  }
  build_weff_kernel<<<(4u * 1024 * 1024) / 256, 256, 0, stream>>>(wa, (unsigned*)weff);

  proj_gemm_kernel<<<dim3(16, 32, 8), 128, 0, stream>>>(xbf, weff, bq, bk, bv, proj);

  size_t topk_lds = 131072 + 4608 + (size_t)128 * 36 * 4;   // 154112 B
  (void)hipFuncSetAttribute(reinterpret_cast<const void*>(topk_kernel),
                            hipFuncAttributeMaxDynamicSharedMemorySize, (int)topk_lds);
  topk_kernel<<<dim3(Bq_ * NHq_, Sq_ / 32), 128, topk_lds, stream>>>(proj, mask, topk_p, kth);

  size_t attn_lds = 73728 + 18432;                          // 92160 B
  (void)hipFuncSetAttribute(reinterpret_cast<const void*>(attn_kernel),
                            hipFuncAttributeMaxDynamicSharedMemorySize, (int)attn_lds);
  attn_kernel<<<dim3(Bq_ * NHq_, Sq_ / 64), 128, attn_lds, stream>>>(proj, mask, kth, (float*)d_out);
}
